// SAE_15710990368942
// MI455X (gfx1250) — matrix operand order repaired and checked by reading, not compiled, not run on hardware
//
#include <hip/hip_runtime.h>
#include <hip/hip_bf16.h>

// SAE topk forward for MI455X (gfx1250, wave32, WMMA, async-LDS DMA).
// B=4096, D_IN=512, HIDDEN=16384, K=32.
// Kernel 0: one-time W_enc f32 -> f16 conversion into workspace.
// Kernel 1: fused encoder GEMM (f16 WMMA, f32 acc) + bias + ReLU + running top-32,
//           W chunks double-buffered into LDS via global_load_async_to_lds_b128.
// Kernel 2: sparse gather decode (K=32 atoms per row) + b_dec.

typedef __attribute__((ext_vector_type(16))) _Float16 v16h;
typedef __attribute__((ext_vector_type(8)))  _Float16 v8h;
typedef __attribute__((ext_vector_type(8)))  float    v8f;

constexpr int BATCH   = 4096;
constexpr int DIN     = 512;
constexpr int HIDDEN  = 16384;
constexpr int KTOP    = 32;

constexpr int ROWS    = 64;          // batch rows per block
constexpr int CHUNK   = 64;          // hidden columns per chunk
constexpr int SX_LD   = DIN + 16;    // padded LDS row (halves); rows stay 32B aligned
constexpr int ROWB    = SX_LD * 2;   // padded LDS row pitch in bytes (1056)
constexpr int NCHUNKS = HIDDEN / CHUNK;   // 256

// ---------------- Kernel 0: W_enc fp32 -> f16 ----------------
__global__ __launch_bounds__(256)
void sae_prep_f16(const float* __restrict__ W_enc, _Float16* __restrict__ wh, int n) {
    for (int i = blockIdx.x * 256 + threadIdx.x; i < n; i += gridDim.x * 256)
        wh[i] = (_Float16)W_enc[i];
}

// ---------------- Kernel 1: fused encode + topk ----------------
template <bool ASYNC>
__global__ __launch_bounds__(256)
void sae_encode_topk(const float* __restrict__ x,
                     const float* __restrict__ W_enc,      // used when !ASYNC
                     const _Float16* __restrict__ wh,      // used when ASYNC
                     const float* __restrict__ b_enc,
                     const float* __restrict__ b_dec,
                     float* __restrict__ topV,
                     int*   __restrict__ topI) {
    __shared__ __align__(16) _Float16 sX[ROWS][SX_LD];      // 67.6 KB
    __shared__ __align__(16) _Float16 sW[2][CHUNK][SX_LD];  // 135.2 KB (double buffer)
    __shared__ float sAct[ROWS][CHUNK + 4];                 // 17.4 KB
    __shared__ float tkVal[ROWS][KTOP];                     // 8 KB
    __shared__ int   tkIdx[ROWS][KTOP];                     // 8 KB

    const int tid  = threadIdx.x;
    const int wave = tid >> 5;              // 0..7
    const int lane = tid & 31;
    const int rowTile  = (wave & 3) * 16;   // 0,16,32,48
    const int colTile  = (wave >> 2) * 32;  // 0 or 32 (two 16-col tiles per wave)
    const int halfSel  = lane >> 4;         // 0 or 1
    const int lane16   = lane & 15;
    const int blockRow = blockIdx.x * ROWS;

    // Async DMA of one f16 W chunk (CHUNK x DIN halves = 64KB) into sW[buf].
    // GVS mode: saddr = uniform 64-bit base, vaddr = per-lane byte offset.
    // LDS dest = low 32 bits of generic shared pointer (= wave-relative LDS offset).
    auto issue_chunk_async = [&](int ch, int buf) {
        const uint64_t gbase   = (uint64_t)(uintptr_t)wh + (uint64_t)ch * (CHUNK * DIN * 2);
        const uint32_t ldsBase = (uint32_t)(uintptr_t)&sW[buf][0][0];
        #pragma unroll
        for (int t = 0; t < (CHUNK * DIN * 2) / (256 * 16); ++t) {  // 16 x b128 per lane
            uint32_t g    = (uint32_t)tid + t * 256;                 // 16B granule index
            uint32_t voff = g << 4;                                  // global byte offset
            uint32_t lds  = ldsBase + (g >> 6) * ROWB + ((g & 63) << 4);
            asm volatile("global_load_async_to_lds_b128 %0, %1, %2"
                         :: "v"(lds), "v"(voff), "s"(gbase) : "memory");
        }
    };
    // Sync fallback: fp32 load + convert + ds_store into sW[buf].
    auto stage_chunk_sync = [&](int ch, int buf) {
        const int base = ch * CHUNK;
        for (int i = tid; i < CHUNK * DIN; i += 256) {
            int r = i >> 9, c = i & (DIN - 1);
            sW[buf][r][c] = (_Float16)W_enc[(size_t)(base + r) * DIN + c];
        }
    };

    // ---- prologue: kick chunk 0, stage x tile, init top-K ----
    if constexpr (ASYNC) issue_chunk_async(0, 0);
    for (int i = tid; i < ROWS * DIN; i += 256) {
        int r = i >> 9, c = i & (DIN - 1);
        sX[r][c] = (_Float16)(x[(size_t)(blockRow + r) * DIN + c] - b_dec[c]);
    }
    if constexpr (!ASYNC) stage_chunk_sync(0, 0);
    if (tid < ROWS) {
        #pragma unroll
        for (int j = 0; j < KTOP; ++j) { tkVal[tid][j] = -1.0f; tkIdx[tid][j] = j; }
    }
    float runMin = -1.0f;   // live only for tid < ROWS
    int   runPos = 0;
    if constexpr (ASYNC) asm volatile("s_wait_asynccnt 0x0" ::: "memory");
    __syncthreads();

    const _Float16* xrow = &sX[rowTile + lane16][0];

    for (int ch = 0; ch < NCHUNKS; ++ch) {
        const int buf = ch & 1;
        // stage next chunk into the other buffer while we compute this one
        if (ch + 1 < NCHUNKS) {
            if constexpr (ASYNC) issue_chunk_async(ch + 1, buf ^ 1);
            else                 stage_chunk_sync(ch + 1, buf ^ 1);
        }

        // ---- two 16x16x512 WMMA tiles per wave, shared A fragment ----
        v8f acc0 = {}, acc1 = {};
        const _Float16* wrow0 = &sW[buf][colTile + lane16][0];
        const _Float16* wrow1 = &sW[buf][colTile + 16 + lane16][0];
        #pragma unroll
        for (int kb = 0; kb < DIN; kb += 32) {
            // A lanes 0-15: K = kb+0..15; lanes 16-31: K = kb+16..31
            const int ka = kb + halfSel * 16;
            v8h a_lo = *(const v8h*)(xrow + ka);
            v8h a_hi = *(const v8h*)(xrow + ka + 8);
            // B lanes 0-15: K = kb+0..15; lanes 16-31: K = kb+16..31
            const int kbb = kb + halfSel * 16;
            v8h b0_lo = *(const v8h*)(wrow0 + kbb);
            v8h b0_hi = *(const v8h*)(wrow0 + kbb + 8);
            v8h b1_lo = *(const v8h*)(wrow1 + kbb);
            v8h b1_hi = *(const v8h*)(wrow1 + kbb + 8);
            v16h af, b0f, b1f;
            #pragma unroll
            for (int i = 0; i < 8; ++i) {
                af[i]  = a_lo[i];  af[i + 8]  = a_hi[i];
                b0f[i] = b0_lo[i]; b0f[i + 8] = b0_hi[i];
                b1f[i] = b1_lo[i]; b1f[i + 8] = b1_hi[i];
            }
            acc0 = __builtin_amdgcn_wmma_f32_16x16x32_f16(
                false, af, false, b0f, (short)0, acc0, false, false);
            acc1 = __builtin_amdgcn_wmma_f32_16x16x32_f16(
                false, af, false, b1f, (short)0, acc1, false, false);
        }

        // ---- bias + ReLU, stage activation tile to LDS ----
        const int chunkBase = ch * CHUNK;
        const float bias0 = b_enc[chunkBase + colTile + lane16];
        const float bias1 = b_enc[chunkBase + colTile + 16 + lane16];
        const int rbase = rowTile + halfSel * 8;
        #pragma unroll
        for (int r = 0; r < 8; ++r) {
            float v0 = acc0[r] + bias0;
            float v1 = acc1[r] + bias1;
            sAct[rbase + r][colTile + lane16]      = v0 > 0.0f ? v0 : 0.0f;
            sAct[rbase + r][colTile + 16 + lane16] = v1 > 0.0f ? v1 : 0.0f;
        }
        __syncthreads();

        // ---- running top-K update: one owner thread per batch row ----
        if (tid < ROWS) {
            #pragma unroll 4
            for (int c = 0; c < CHUNK; ++c) {
                float v = sAct[tid][c];
                if (v > runMin) {
                    tkVal[tid][runPos] = v;
                    tkIdx[tid][runPos] = chunkBase + c;
                    float mv = tkVal[tid][0]; int mp = 0;
                    #pragma unroll
                    for (int j = 1; j < KTOP; ++j) {
                        float t = tkVal[tid][j];
                        if (t < mv) { mv = t; mp = j; }
                    }
                    runMin = mv; runPos = mp;
                }
            }
        }
        // drain this iteration's async DMA before the buffer-swap barrier
        if constexpr (ASYNC) asm volatile("s_wait_asynccnt 0x0" ::: "memory");
        __syncthreads();
    }

    // ---- emit top-K lists ----
    if (tid < ROWS) {
        const int gr = blockRow + tid;
        #pragma unroll
        for (int j = 0; j < KTOP; ++j) {
            float v = tkVal[tid][j];
            topV[gr * KTOP + j] = v > 0.0f ? v : 0.0f;
            topI[gr * KTOP + j] = tkIdx[tid][j];
        }
    }
}

// ---------------- Kernel 2: sparse decode ----------------
__global__ __launch_bounds__(256)
void sae_decode(const float* __restrict__ topV,
                const int*   __restrict__ topI,
                const float* __restrict__ W_dec,
                const float* __restrict__ b_dec,
                float* __restrict__ out) {
    const int row = blockIdx.x;
    const int tid = threadIdx.x;
    __shared__ float sv[KTOP];
    __shared__ int   si[KTOP];
    if (tid < KTOP) {
        sv[tid] = topV[row * KTOP + tid];
        si[tid] = topI[row * KTOP + tid];
    }
    __syncthreads();

    const int c0 = tid;            // cols tid and tid+256
    float acc0 = b_dec[c0];
    float acc1 = b_dec[c0 + 256];
    #pragma unroll 8
    for (int j = 0; j < KTOP; ++j) {
        const float v = sv[j];
        const float* wr = W_dec + (size_t)si[j] * DIN;
        acc0 += v * wr[c0];
        acc1 += v * wr[c0 + 256];
    }
    out[(size_t)row * DIN + c0]       = acc0;
    out[(size_t)row * DIN + c0 + 256] = acc1;
}

// ---------------- launcher ----------------
extern "C" void kernel_launch(void* const* d_in, const int* in_sizes, int n_in,
                              void* d_out, int out_size, void* d_ws, size_t ws_size,
                              hipStream_t stream) {
    const float* x     = (const float*)d_in[0];
    const float* W_enc = (const float*)d_in[1];
    const float* b_enc = (const float*)d_in[2];
    const float* W_dec = (const float*)d_in[3];
    const float* b_dec = (const float*)d_in[4];
    // d_in[5] is k (==32), hardcoded as KTOP.

    float*    topV = (float*)d_ws;                                      // 512 KB
    int*      topI = (int*)((char*)d_ws + BATCH * KTOP * sizeof(float)); // +512 KB
    _Float16* wh   = (_Float16*)((char*)d_ws + (1u << 20));              // 16 MB f16 W_enc

    const size_t need = (1u << 20) + (size_t)HIDDEN * DIN * sizeof(_Float16);
    if (ws_size >= need) {
        sae_prep_f16<<<4096, 256, 0, stream>>>(W_enc, wh, HIDDEN * DIN);
        sae_encode_topk<true><<<BATCH / ROWS, 256, 0, stream>>>(
            x, W_enc, wh, b_enc, b_dec, topV, topI);
    } else {
        sae_encode_topk<false><<<BATCH / ROWS, 256, 0, stream>>>(
            x, W_enc, nullptr, b_enc, b_dec, topV, topI);
    }
    sae_decode<<<BATCH, 256, 0, stream>>>(topV, topI, W_dec, b_dec, (float*)d_out);
}
